// Conv2DWeightDemod_14946486190095
// MI455X (gfx1250) — matrix operand order repaired and checked by reading, not compiled, not run on hardware
//
#include <hip/hip_runtime.h>

#define BATCH 8
#define CIN   256
#define COUT  256
#define KSZ   3
#define HH    128
#define WW    128
#define HWPIX (HH*WW)            // 16384
#define KGEMM (CIN*KSZ*KSZ)      // 2304
#define NSTEP (KGEMM/32)         // 72

typedef _Float16 v16h __attribute__((ext_vector_type(16)));
typedef float    v8f  __attribute__((ext_vector_type(8)));

union Frag {
    v16h h;
    unsigned int u[8];
};

// ---------------- kernel 1a: wsq[i][kw] = sum_{o,kh} w[o,i,kh,kw]^2 ----------
__global__ __launch_bounds__(256)
void k_wsq(const float* __restrict__ w, float* __restrict__ wsq) {
    int idx = blockIdx.x * blockDim.x + threadIdx.x;
    if (idx >= CIN * KSZ) return;
    int i = idx / KSZ, kw = idx % KSZ;
    float s = 0.f;
    for (int o = 0; o < COUT; ++o) {
        const float* wp = w + ((size_t)o * CIN + i) * (KSZ * KSZ) + kw;
        s += wp[0] * wp[0] + wp[3] * wp[3] + wp[6] * wp[6];
    }
    wsq[idx] = s;
}

// ---------------- kernel 1b: demod[b][kw] = rsqrt(sum_i s^2 * wsq + 1e-8) ----
__global__ __launch_bounds__(32)
void k_demod(const float* __restrict__ style, const float* __restrict__ wsq,
             float* __restrict__ demod) {
    int idx = threadIdx.x;
    if (idx >= BATCH * KSZ) return;
    int bb = idx / KSZ, kw = idx % KSZ;
    float s = 0.f;
    for (int i = 0; i < CIN; ++i) {
        float sv = style[bb * CIN + i];
        s += sv * sv * wsq[i * KSZ + kw];
    }
    demod[idx] = rsqrtf(s + 1e-8f);
}

// ------------- kernel 1c: wm[b][o][k] (f16), k = (kh*3+kw)*256 + cin ---------
__global__ __launch_bounds__(256)
void k_modw(const float* __restrict__ w, const float* __restrict__ style,
            const float* __restrict__ demod, _Float16* __restrict__ wm) {
    int idx = blockIdx.x * 256 + threadIdx.x;        // < BATCH*COUT*KGEMM
    int k   = idx % KGEMM;
    int o   = (idx / KGEMM) % COUT;
    int bb  = idx / (KGEMM * COUT);
    int q   = k >> 8;                 // kh*3 + kw  (0..8)
    int cin = k & 255;
    int kh  = (q * 11) >> 5;          // q/3 for q in [0,8]
    int kw  = q - kh * 3;
    float v = w[((size_t)o * CIN + cin) * 9 + kh * 3 + kw]
              * style[bb * CIN + cin] * demod[bb * 3 + kw];
    wm[(size_t)idx] = (_Float16)v;    // idx == ((bb*COUT+o)*KGEMM)+k
}

// ---------------- kernel 2: implicit-GEMM conv + leaky ReLU ------------------
// Block: 256 thr (8 waves). Tile: 128 couts x 128 pixels (one image row).
// Wave: 32(m) x 64(n) -> 8 WMMA/step. Double-buffered LDS; A tile via
// global_load_async_to_lds_b64 (ASYNCcnt); B gather pointer hoisted per tap.
__global__ __launch_bounds__(256)
void k_conv(const float* __restrict__ x, const _Float16* __restrict__ wm,
            float* __restrict__ out) {
    __shared__ _Float16 Ash[2][128][36];   // [buf][m][k]
    __shared__ _Float16 Bsh[2][128][36];   // [buf][n][k] (transposed im2col)

    const int tid  = threadIdx.x;
    const int lane = tid & 31;
    const int wave = tid >> 5;
    const int hi   = lane >> 4;
    const int ln   = lane & 15;

    const int bb   = blockIdx.z;
    const int m0   = blockIdx.y * 128;
    const int hrow = blockIdx.x;           // pixel tile == one image row

    const int wm0 = (wave >> 1) * 32;      // 4 m-waves
    const int wn0 = (wave & 1) * 64;       // 2 n-waves x 64 px

    const _Float16* Abase = wm + ((size_t)bb * COUT + m0) * KGEMM;
    const float*    Xbase = x + (size_t)bb * CIN * HWPIX;

    // ---- per-thread constant mapping for the A async copy (4 x b64/lane) ----
    int arow[4], acol[4];
    #pragma unroll
    for (int it = 0; it < 4; ++it) {
        int chunk = tid + it * 256;        // 0..1023
        arow[it] = chunk >> 3;             // 0..127
        acol[it] = (chunk & 7) << 2;       // 0,4,...,28 halves
    }
    // ---- per-thread constants for the B gather ----
    const int bc  = tid & 127;             // pixel (column) within the row
    const int bp0 = tid >> 7;              // 0..1 (pair group)

    auto issue_A = [&](int buf, int s) {
        #pragma unroll
        for (int it = 0; it < 4; ++it) {
            const _Float16* g = Abase + (size_t)arow[it] * KGEMM + s * 32 + acol[it];
            unsigned lds = (unsigned)(uintptr_t)&Ash[buf][arow[it]][acol[it]];
            asm volatile("global_load_async_to_lds_b64 %0, %1, off"
                         :: "v"(lds), "v"(g) : "memory");
        }
    };

    // ---- persistent B gather state (recomputed only when the tap changes) ---
    const float* xp = nullptr;             // base for this thread, current step
    bool okb = false;
    auto setup_B = [&](int s) {            // s is a multiple of 8 (cin0 == 0)
        int q   = s >> 3;                  // tap 0..8
        int kh  = (q * 11) >> 5;           // q/3
        int kw  = q - kh * 3;
        int hs  = hrow + kh - 1;
        int wsx = bc + kw - 1;
        okb = (hs >= 0) & (hs < HH) & (wsx >= 0) & (wsx < WW);
        int hc = min(max(hs, 0), HH - 1);
        int wc = min(max(wsx, 0), WW - 1);
        xp = Xbase + (size_t)(2 * bp0) * HWPIX + hc * WW + wc;
    };

    auto load_B = [&](float* reg) {        // 16 values: pairs pr = bp0 + 2*i
        const float* p = xp;
        #pragma unroll
        for (int i = 0; i < 8; ++i) {
            float a = p[0];
            float b = p[HWPIX];
            reg[2 * i]     = okb ? a : 0.f;   // cndmask, no exec divergence
            reg[2 * i + 1] = okb ? b : 0.f;
            p += 4 * (size_t)HWPIX;           // next pair group: +4 cin
        }
    };

    auto store_B = [&](int buf, const float* reg) {
        #pragma unroll
        for (int i = 0; i < 8; ++i) {
            int r = 2 * (bp0 + 2 * i);        // even k index
            union { _Float16 h[2]; unsigned u; } pk;
            pk.h[0] = (_Float16)reg[2 * i];
            pk.h[1] = (_Float16)reg[2 * i + 1];
            *(unsigned*)&Bsh[buf][bc][r] = pk.u;
        }
    };

    v8f acc[2][4];
    #pragma unroll
    for (int mi = 0; mi < 2; ++mi)
        #pragma unroll
        for (int nf = 0; nf < 4; ++nf)
            acc[mi][nf] = (v8f){0.f,0.f,0.f,0.f,0.f,0.f,0.f,0.f};

    // ---- prologue: fill buffer 0 ----
    float breg[16];
    issue_A(0, 0);
    setup_B(0);
    load_B(breg);
    store_B(0, breg);
    asm volatile("s_wait_asynccnt 0x0" ::: "memory");
    __syncthreads();

    for (int s = 0; s < NSTEP; ++s) {
        const int cur = s & 1;
        if (s + 1 < NSTEP) {               // uniform branch, EXEC stays full
            issue_A(cur ^ 1, s + 1);
            if (((s + 1) & 7) == 0) setup_B(s + 1);   // tap change (uniform)
            else                    xp += (size_t)32 * HWPIX;
            load_B(breg);
        }

        // ---- fragments per ISA 16-bit A(16x32)/B(32x16) layouts ----
        Frag a0, a1, bfr[4];
        #pragma unroll
        for (int vv = 0; vv < 8; ++vv) {
            int kb  = ((vv & 3) << 1) + ((vv >> 2) << 4) + (hi << 3);
            a0.u[vv] = *(const unsigned*)&Ash[cur][wm0 + ln][kb];
            a1.u[vv] = *(const unsigned*)&Ash[cur][wm0 + 16 + ln][kb];
            int kb2 = ((vv & 3) << 1) + ((vv >> 2) << 4) + (hi << 3);
            #pragma unroll
            for (int nf = 0; nf < 4; ++nf)
                bfr[nf].u[vv] = *(const unsigned*)&Bsh[cur][wn0 + nf * 16 + ln][kb2];
        }
        #pragma unroll
        for (int nf = 0; nf < 4; ++nf) {
            acc[0][nf] = __builtin_amdgcn_wmma_f32_16x16x32_f16(
                false, a0.h, false, bfr[nf].h, (short)0, acc[0][nf], false, false);
            acc[1][nf] = __builtin_amdgcn_wmma_f32_16x16x32_f16(
                false, a1.h, false, bfr[nf].h, (short)0, acc[1][nf], false, false);
        }

        if (s + 1 < NSTEP) {
            store_B(cur ^ 1, breg);
            asm volatile("s_wait_asynccnt 0x0" ::: "memory");
        }
        __syncthreads();
    }

    // ---- epilogue: C/D layout -> leaky ReLU -> fp32 store ----
    float* obase = out + (size_t)bb * COUT * HWPIX + (size_t)hrow * WW;
    #pragma unroll
    for (int mi = 0; mi < 2; ++mi) {
        #pragma unroll
        for (int j = 0; j < 8; ++j) {
            int mg = m0 + wm0 + mi * 16 + j + (hi << 3);
            float* orow = obase + (size_t)mg * HWPIX;
            #pragma unroll
            for (int nf = 0; nf < 4; ++nf) {
                int ng = wn0 + nf * 16 + ln;
                float v = acc[mi][nf][j];
                orow[ng] = v > 0.f ? v : 0.2f * v;
            }
        }
    }
}

extern "C" void kernel_launch(void* const* d_in, const int* in_sizes, int n_in,
                              void* d_out, int out_size, void* d_ws, size_t ws_size,
                              hipStream_t stream) {
    (void)in_sizes; (void)n_in; (void)out_size; (void)ws_size;
    const float* x     = (const float*)d_in[0];   // (B,CIN,H,W)
    const float* style = (const float*)d_in[1];   // (B,CIN)
    const float* w     = (const float*)d_in[2];   // (COUT,CIN,3,3)
    float* out = (float*)d_out;

    char* ws        = (char*)d_ws;
    float* wsq      = (float*)ws;                 // 768 floats
    float* demod    = (float*)(ws + 4096);        // 24 floats
    _Float16* wmbuf = (_Float16*)(ws + 8192);     // 8*256*2304 halves (~9.4 MB)

    k_wsq  <<<3, 256, 0, stream>>>(w, wsq);
    k_demod<<<1, 32,  0, stream>>>(style, wsq, demod);

    int total = BATCH * COUT * KGEMM;             // 4,718,592
    k_modw <<<total / 256, 256, 0, stream>>>(w, style, demod, wmbuf);

    dim3 grid(HH, COUT / 128, BATCH);             // 128 x 2 x 8 blocks
    k_conv <<<grid, 256, 0, stream>>>(x, wmbuf, out);
}
